// LocalAttn_56092272885960
// MI455X (gfx1250) — matrix operand order repaired and checked by reading, not compiled, not run on hardware
//
#include <hip/hip_runtime.h>
#include <hip/hip_bf16.h>
#include <math.h>

// ---------------------------------------------------------------------------
// Local-attention refinement head for MI455X (gfx1250, wave32, WMMA).
// All GEMM-shaped compute -> v_wmma_f32_16x16x32_f16.
// K/V projections hoisted out of the 5x5 unfold (project map once, add
// (pos+emb)@W^T+b as a 25x64 table).
// Data movement: float4 global loads -> packed f16x4 LDS stores -> ds_load_b128
// fragment reads; global_prefetch for the next k-step.
// ---------------------------------------------------------------------------

typedef __attribute__((ext_vector_type(16))) _Float16 v16h;
typedef __attribute__((ext_vector_type(4)))  _Float16 v4h;
typedef __attribute__((ext_vector_type(8)))  float    v8f;
typedef __attribute__((ext_vector_type(4)))  float    v4f;

#define KSTEP 32

// --- WMMA fragment loaders (per CDNA5 ISA 7.12.2 layouts) -------------------
// A: 16-bit 16x32 (MxK). lane%16 = M; lane/16 picks K half; per-lane data is
// halves [kb..kb+7] and [kb+16..kb+23] -> two ds_load_b128.
__device__ __forceinline__ v16h frag_a(const _Float16* p0, int ldk) {
  const int lane = threadIdx.x & 31;
  const int row  = lane & 15;
  const int kb   = (lane >> 4) << 3;          // 0 or 8
  const _Float16* p = p0 + row * ldk + kb;
  v16h f;
#pragma unroll
  for (int r = 0; r < 8; ++r) {
    const int k = ((r & 3) << 1) + ((r >> 2) << 4);
    f[2 * r]     = p[k];
    f[2 * r + 1] = p[k + 1];
  }
  return f;
}

// B: 32x16 (KxN), sourced from row-major W[N][K] (i.e. B^T): lane%16 = N,
// lane/16 picks K half; per-lane data is [kb..kb+7] and [kb+16..kb+23] (same as A).
__device__ __forceinline__ v16h frag_b(const _Float16* p0, int ldk) {
  const int lane = threadIdx.x & 31;
  const int col  = lane & 15;
  const int kb   = (lane >> 4) << 3;          // 0 or 8
  const _Float16* p = p0 + col * ldk + kb;
  v16h f;
#pragma unroll
  for (int r = 0; r < 8; ++r) {
    const int k = ((r & 3) << 1) + ((r >> 2) << 4);
    f[2 * r]     = p[k];
    f[2 * r + 1] = p[k + 1];
  }
  return f;
}

__device__ __forceinline__ v4h cvt4(const v4f v) {
  v4h h;
  h[0] = (_Float16)v[0]; h[1] = (_Float16)v[1];
  h[2] = (_Float16)v[2]; h[3] = (_Float16)v[3];
  return h;
}

// ---------------------------------------------------------------------------
// Generic WMMA GEMM:  out[M,N] = act(A[M,K] @ W[N,K]^T + bias[N]) + res[M,N]
// block = 256 threads (8 waves), C tile = 32(M) x 64(N); wave (mt,nt) does one
// 16x16 subtile. grid = (M/32, N/64).  act: 0=none, 1=relu, 2=gelu(exact)
// ---------------------------------------------------------------------------
__global__ void gemm_wmma_kernel(const float* __restrict__ A,
                                 const float* __restrict__ W,
                                 const float* __restrict__ bias,
                                 const float* __restrict__ res,
                                 float* __restrict__ out,
                                 int M, int N, int K, int act) {
  __shared__ __align__(16) _Float16 Ah[32 * KSTEP];
  __shared__ __align__(16) _Float16 Wh[64 * KSTEP];
  const int tid  = threadIdx.x;
  const int wv   = tid >> 5;
  const int lane = tid & 31;
  const int mb0  = blockIdx.x << 5;
  const int nb0  = blockIdx.y << 6;
  const int mt   = (wv >> 2) << 4;            // 0 or 16
  const int nt   = (wv & 3) << 4;             // 0,16,32,48

  v8f acc = {};
  for (int k0 = 0; k0 < K; k0 += KSTEP) {
    // A tile 32x32: one float4 per thread
    {
      const int mi = tid >> 3, kc = (tid & 7) << 2;
      const v4f av = *(const v4f*)(A + (size_t)(mb0 + mi) * K + k0 + kc);
      *(v4h*)(Ah + mi * KSTEP + kc) = cvt4(av);
      if (k0 + KSTEP < K)
        __builtin_prefetch(A + (size_t)(mb0 + mi) * K + k0 + KSTEP + kc, 0, 0);
    }
    // W tile 64x32: two float4 per thread
#pragma unroll
    for (int h = 0; h < 2; ++h) {
      const int i  = tid + h * 256;
      const int ni = i >> 3, kc = (i & 7) << 2;
      const v4f wvv = *(const v4f*)(W + (size_t)(nb0 + ni) * K + k0 + kc);
      *(v4h*)(Wh + ni * KSTEP + kc) = cvt4(wvv);
      if (k0 + KSTEP < K)
        __builtin_prefetch(W + (size_t)(nb0 + ni) * K + k0 + KSTEP + kc, 0, 0);
    }
    __syncthreads();
    v16h a  = frag_a(Ah + mt * KSTEP, KSTEP);
    v16h bf = frag_b(Wh + nt * KSTEP, KSTEP);
    acc = __builtin_amdgcn_wmma_f32_16x16x32_f16(false, a, false, bf,
                                                 (short)0, acc, false, false);
    __syncthreads();
  }

  const int nl = lane & 15;
  const int mb = (lane >> 4) << 3;
  const int n  = nb0 + nt + nl;
#pragma unroll
  for (int r = 0; r < 8; ++r) {
    const int m = mb0 + mt + mb + r;
    float v = acc[r];
    if (bias) v += bias[n];
    if (act == 1)      v = fmaxf(v, 0.f);
    else if (act == 2) v = 0.5f * v * (1.f + erff(v * 0.70710678118f));
    if (res) v += res[(size_t)m * N + n];
    out[(size_t)m * N + n] = v;
  }
}

// ---------------------------------------------------------------------------
// RCU conv3x3 (64->64) as implicit GEMM. 9 taps x 2 k-steps of WMMA.
// out_rows[pix][64] = conv3x3(relu(x)) + b (+ res). Input/res via strides.
// block = 128 threads (4 waves); one block = 16 pixels (contiguous in W) x 64 oc.
// ---------------------------------------------------------------------------
__global__ void rcu_conv_kernel(const float* __restrict__ x,
                                long sb, long sc, long sy, long sx,
                                const float* __restrict__ w,
                                const float* __restrict__ bias,
                                const float* __restrict__ res,
                                long rb, long rc, long ry, long rx,
                                float* __restrict__ out,
                                int Bn, int H, int W) {
  __shared__ __align__(16) _Float16 Ah[16 * 64];
  __shared__ __align__(16) _Float16 Wh[64 * 64];
  const int tid  = threadIdx.x;
  const int wv   = tid >> 5;
  const int lane = tid & 31;

  const int tilesPerRow = W >> 4;
  int t = blockIdx.x;
  const int x0 = (t % tilesPerRow) << 4; t /= tilesPerRow;
  const int y  = t % H;
  const int b  = t / H;

  v8f acc = {};
#pragma unroll 1
  for (int tap = 0; tap < 9; ++tap) {
    const int dy = tap / 3 - 1, dx = tap % 3 - 1;
    const int yy = y + dy;
    // activations: 16 px x 16 channel-quads, packed f16x4 LDS stores
#pragma unroll
    for (int h = 0; h < 2; ++h) {
      const int i  = tid + h * 128;
      const int pi = i >> 4, cg = (i & 15) << 2;
      const int xx = x0 + pi + dx;
      v4h hv;
      if (yy >= 0 && yy < H && xx >= 0 && xx < W) {
        const float* px = x + (long)b * sb + (long)yy * sy + (long)xx * sx;
#pragma unroll
        for (int j = 0; j < 4; ++j)
          hv[j] = (_Float16)fmaxf(px[(long)(cg + j) * sc], 0.f);
      } else {
        hv[0] = hv[1] = hv[2] = hv[3] = (_Float16)0.f;
      }
      *(v4h*)(Ah + pi * 64 + cg) = hv;
    }
    // weights for this tap: 64 oc x 16 ic-quads
#pragma unroll
    for (int h = 0; h < 8; ++h) {
      const int i  = tid + h * 128;
      const int oc = i >> 4, icg = (i & 15) << 2;
      v4h hv;
#pragma unroll
      for (int j = 0; j < 4; ++j)
        hv[j] = (_Float16)w[((oc * 64 + icg + j) * 3 + (dy + 1)) * 3 + (dx + 1)];
      *(v4h*)(Wh + oc * 64 + icg) = hv;
    }
    __syncthreads();
#pragma unroll
    for (int ks = 0; ks < 64; ks += KSTEP) {
      v16h a  = frag_a(Ah + ks, 64);
      v16h bf = frag_b(Wh + (wv << 4) * 64 + ks, 64);
      acc = __builtin_amdgcn_wmma_f32_16x16x32_f16(false, a, false, bf,
                                                   (short)0, acc, false, false);
    }
    __syncthreads();
  }

  const int nl = lane & 15;
  const int mb = (lane >> 4) << 3;
  const int oc = (wv << 4) + nl;
  const long pixBase = ((long)b * H + y) * W + x0;
#pragma unroll
  for (int r = 0; r < 8; ++r) {
    const int pi = mb + r;
    float v = acc[r] + bias[oc];
    if (res)
      v += res[(long)b * rb + (long)oc * rc + (long)y * ry + (long)(x0 + pi) * rx];
    out[(pixBase + pi) * 64 + oc] = v;
  }
}

// --- LayerNorm over D=64, one wave32 per token ------------------------------
__global__ void ln_kernel(const float* __restrict__ x,
                          const float* __restrict__ g,
                          const float* __restrict__ b,
                          float* __restrict__ out, int M) {
  const int tid  = threadIdx.x;
  const int wid  = (blockIdx.x * blockDim.x + tid) >> 5;
  const int lane = tid & 31;
  if (wid >= M) return;
  const size_t base = (size_t)wid * 64;
  float v0 = x[base + lane], v1 = x[base + 32 + lane];
  float s = v0 + v1;
#pragma unroll
  for (int o = 16; o; o >>= 1) s += __shfl_xor(s, o, 32);
  const float mean = s * (1.f / 64.f);
  const float d0 = v0 - mean, d1 = v1 - mean;
  float q = d0 * d0 + d1 * d1;
#pragma unroll
  for (int o = 16; o; o >>= 1) q += __shfl_xor(q, o, 32);
  const float inv = rsqrtf(q * (1.f / 64.f) + 1e-5f);
  out[base + lane]      = d0 * inv * g[lane]      + b[lane];
  out[base + 32 + lane] = d1 * inv * g[lane + 32] + b[lane + 32];
}

// --- kpos/vpos: out[j][n] = sum_k (pos[j][k]+emb[k]) * W[n][k] + bias[n] ----
__global__ void posproj_kernel(const float* __restrict__ pos,
                               const float* __restrict__ emb,
                               const float* __restrict__ W,
                               const float* __restrict__ bias,
                               float* __restrict__ outp) {
  const int j = blockIdx.x, n = threadIdx.x;
  float s = bias[n];
#pragma unroll
  for (int k = 0; k < 64; ++k) s += (pos[j * 64 + k] + emb[k]) * W[n * 64 + k];
  outp[j * 64 + n] = s;
}

// ---------------------------------------------------------------------------
// 25-neighbor attention, one wave32 per token. lane -> 2 channels; heads are
// aligned 8-lane groups -> shfl_xor(1,2,4) reduces q.k per head.
// K[n,j] = Kmap[nbr] + kpos[j]   (zero-padded unfold => kpos only outside).
// ---------------------------------------------------------------------------
__global__ void attn_kernel(const float* __restrict__ Q,
                            const float* __restrict__ Km,
                            const float* __restrict__ Vm,
                            const float* __restrict__ kpos,
                            const float* __restrict__ vpos,
                            float* __restrict__ out,
                            int Bn, int H, int W) {
  const int tid  = threadIdx.x;
  const int wid  = (blockIdx.x * blockDim.x + tid) >> 5;
  const int lane = tid & 31;
  const int M = Bn * H * W;
  if (wid >= M) return;
  const int xw = wid % W;
  const int yw = (wid / W) % H;
  const int b  = wid / (W * H);
  const int d0 = lane * 2, d1 = d0 + 1;
  const float q0 = Q[(size_t)wid * 64 + d0];
  const float q1 = Q[(size_t)wid * 64 + d1];

  float lg[25];
#pragma unroll
  for (int j = 0; j < 25; ++j) {
    const int dy = j / 5 - 2, dx = j % 5 - 2;
    const int yy = yw + dy, xx = xw + dx;
    const bool valid = (yy >= 0 && yy < H && xx >= 0 && xx < W);
    float k0 = kpos[j * 64 + d0], k1 = kpos[j * 64 + d1];
    if (valid) {
      const size_t nb = ((size_t)b * H + yy) * W + xx;
      k0 += Km[nb * 64 + d0];
      k1 += Km[nb * 64 + d1];
    }
    float p = q0 * k0 + q1 * k1;
    p += __shfl_xor(p, 1, 32);
    p += __shfl_xor(p, 2, 32);
    p += __shfl_xor(p, 4, 32);
    lg[j] = p * 0.25f;                       // 1/sqrt(16)
  }
  float mx = -1e30f;
#pragma unroll
  for (int j = 0; j < 25; ++j) mx = fmaxf(mx, lg[j]);
  float ssum = 0.f;
#pragma unroll
  for (int j = 0; j < 25; ++j) { lg[j] = expf(lg[j] - mx); ssum += lg[j]; }
  const float inv = 1.f / ssum;

  float o0 = 0.f, o1 = 0.f;
#pragma unroll
  for (int j = 0; j < 25; ++j) {
    const int dy = j / 5 - 2, dx = j % 5 - 2;
    const int yy = yw + dy, xx = xw + dx;
    const bool valid = (yy >= 0 && yy < H && xx >= 0 && xx < W);
    float v0 = vpos[j * 64 + d0], v1 = vpos[j * 64 + d1];
    if (valid) {
      const size_t nb = ((size_t)b * H + yy) * W + xx;
      v0 += Vm[nb * 64 + d0];
      v1 += Vm[nb * 64 + d1];
    }
    const float wj = lg[j] * inv;
    o0 += wj * v0;
    o1 += wj * v1;
  }
  out[(size_t)wid * 64 + d0] = o0;
  out[(size_t)wid * 64 + d1] = o1;
}

// --- bilinear resize (align_corners=True) on [B,Hi,Wi,C] rows ---------------
__global__ void resize_rows_kernel(const float* __restrict__ in,
                                   float* __restrict__ out,
                                   int Bn, int Hi, int Wi, int Ho, int Wo, int C) {
  const long idx = (long)blockIdx.x * blockDim.x + threadIdx.x;
  const long total = (long)Bn * Ho * Wo * C;
  if (idx >= total) return;
  const int c = idx % C;
  long t = idx / C;
  const int xo = t % Wo; t /= Wo;
  const int yo = t % Ho;
  const int b  = t / Ho;
  const float sy = (float)(Hi - 1) / (float)(Ho - 1);
  const float sx = (float)(Wi - 1) / (float)(Wo - 1);
  const float fy = yo * sy;
  const float fx = xo * sx;
  int y0 = (int)floorf(fy); if (y0 > Hi - 2) y0 = Hi - 2; if (y0 < 0) y0 = 0;
  int x0 = (int)floorf(fx); if (x0 > Wi - 2) x0 = Wi - 2; if (x0 < 0) x0 = 0;
  const float wy = fy - y0, wx = fx - x0;
  const size_t rb = ((size_t)b * Hi + y0) * Wi;
  const float a00 = in[(rb + x0) * C + c];
  const float a01 = in[(rb + x0 + 1) * C + c];
  const float a10 = in[(rb + Wi + x0) * C + c];
  const float a11 = in[(rb + Wi + x0 + 1) * C + c];
  const float top = a00 * (1.f - wx) + a01 * wx;
  const float bot = a10 * (1.f - wx) + a11 * wx;
  out[idx] = top * (1.f - wy) + bot * wy;
}

// --- head layer 2: y = softplus(h[M,128] . w2 + b2), one wave per token -----
__global__ void head_final_kernel(const float* __restrict__ h,
                                  const float* __restrict__ w2,
                                  const float* __restrict__ b2,
                                  float* __restrict__ y, int M) {
  const int tid  = threadIdx.x;
  const int wid  = (blockIdx.x * blockDim.x + tid) >> 5;
  const int lane = tid & 31;
  if (wid >= M) return;
  float s = 0.f;
#pragma unroll
  for (int i = 0; i < 4; ++i) {
    const int k = lane + i * 32;
    s += h[(size_t)wid * 128 + k] * w2[k];
  }
#pragma unroll
  for (int o = 16; o; o >>= 1) s += __shfl_xor(s, o, 32);
  if (lane == 0) {
    float v = s + b2[0];
    y[wid] = (v > 20.f) ? v : log1pf(expf(v));
  }
}

// ---------------------------------------------------------------------------
// Host side
// ---------------------------------------------------------------------------
namespace {

struct AttnP {
  const float *in_w, *in_b, *out_w, *out_b;
  const float *ln1_w, *ln1_b, *ln2_w, *ln2_b;
  const float *f1w, *f1b, *f2w, *f2b;
};

struct Scratch {
  float *S1, *S2, *Km, *Vm, *MID, *kpos, *vpos;
};

inline AttnP load_attn(void* const* d_in, int base) {
  AttnP p;
  p.in_w  = (const float*)d_in[base + 0];
  p.in_b  = (const float*)d_in[base + 1];
  p.out_w = (const float*)d_in[base + 2];
  p.out_b = (const float*)d_in[base + 3];
  p.ln1_w = (const float*)d_in[base + 4];
  p.ln1_b = (const float*)d_in[base + 5];
  p.ln2_w = (const float*)d_in[base + 6];
  p.ln2_b = (const float*)d_in[base + 7];
  p.f1w   = (const float*)d_in[base + 8];
  p.f1b   = (const float*)d_in[base + 9];
  p.f2w   = (const float*)d_in[base + 10];
  p.f2b   = (const float*)d_in[base + 11];
  return p;
}

void rcu(const float* x_nchw, const float* w1, const float* b1,
         const float* w2, const float* b2,
         float* mid_rows, float* out_rows,
         int Bn, int H, int W, hipStream_t s) {
  const int tiles = Bn * H * (W >> 4);
  // h1 = conv(relu(x), w1) + b1           (NCHW in -> rows out)
  rcu_conv_kernel<<<tiles, 128, 0, s>>>(
      x_nchw, (long)64 * H * W, (long)H * W, (long)W, 1L,
      w1, b1, nullptr, 0, 0, 0, 0, mid_rows, Bn, H, W);
  // out = x + conv(relu(h1), w2) + b2     (rows in, NCHW residual)
  rcu_conv_kernel<<<tiles, 128, 0, s>>>(
      mid_rows, (long)H * W * 64, 1L, (long)W * 64, 64L,
      w2, b2, x_nchw, (long)64 * H * W, (long)H * W, (long)W, 1L,
      out_rows, Bn, H, W);
}

void cross_block(const float* feat, int Bn, int H, int W, float* X,
                 const AttnP& p, const float* pos, const float* emb_row,
                 const Scratch& ws, hipStream_t s) {
  const int M = Bn * H * W;
  const dim3 g1(M / 32, 1), g4(M / 32, 4);
  // 25x64 pos/emb projections with K/V bias folded in
  posproj_kernel<<<25, 64, 0, s>>>(pos, emb_row, p.in_w + 64 * 64,  p.in_b + 64,  ws.kpos);
  posproj_kernel<<<25, 64, 0, s>>>(pos, emb_row, p.in_w + 128 * 64, p.in_b + 128, ws.vpos);
  // per-pixel K/V maps (unfold factored out)
  gemm_wmma_kernel<<<g1, 256, 0, s>>>(feat, p.in_w + 64 * 64,  nullptr, nullptr, ws.Km, M, 64, 64, 0);
  gemm_wmma_kernel<<<g1, 256, 0, s>>>(feat, p.in_w + 128 * 64, nullptr, nullptr, ws.Vm, M, 64, 64, 0);
  // z = LN1(x); q = z @ Wq^T + bq
  ln_kernel<<<M / 8, 256, 0, s>>>(X, p.ln1_w, p.ln1_b, ws.S1, M);
  gemm_wmma_kernel<<<g1, 256, 0, s>>>(ws.S1, p.in_w, p.in_b, nullptr, ws.S2, M, 64, 64, 0);
  // attention over 25 neighbors
  attn_kernel<<<M / 8, 256, 0, s>>>(ws.S2, ws.Km, ws.Vm, ws.kpos, ws.vpos, ws.S1, Bn, H, W);
  // x = x + O @ Wout^T + bout
  gemm_wmma_kernel<<<g1, 256, 0, s>>>(ws.S1, p.out_w, p.out_b, X, X, M, 64, 64, 0);
  // x = LN2(x) (in place); x = x + gelu(x@W1^T+b1)@W2^T+b2
  ln_kernel<<<M / 8, 256, 0, s>>>(X, p.ln2_w, p.ln2_b, X, M);
  gemm_wmma_kernel<<<g4, 256, 0, s>>>(X, p.f1w, p.f1b, nullptr, ws.MID, M, 256, 64, 2);
  gemm_wmma_kernel<<<g1, 256, 0, s>>>(ws.MID, p.f2w, p.f2b, X, X, M, 64, 256, 0);
}

} // namespace

extern "C" void kernel_launch(void* const* d_in, const int* in_sizes, int n_in,
                              void* d_out, int out_size, void* d_ws, size_t ws_size,
                              hipStream_t stream) {
  (void)in_sizes; (void)n_in; (void)out_size; (void)ws_size;
  const int Bn = 4, H8 = 48, W8 = 64, H4 = 96, W4 = 128;
  const int N8 = Bn * H8 * W8;     // 12288
  const int N4 = Bn * H4 * W4;     // 49152

  // -------- inputs (setup_inputs dict order, nested dicts depth-first) -----
  const float* s8 = (const float*)d_in[0];
  const float* L2 = (const float*)d_in[1];
  const float* L1 = (const float*)d_in[2];
  const float* s8r_w1 = (const float*)d_in[3];
  const float* s8r_b1 = (const float*)d_in[4];
  const float* s8r_w2 = (const float*)d_in[5];
  const float* s8r_b2 = (const float*)d_in[6];
  const float* l2r_w1 = (const float*)d_in[7];
  const float* l2r_b1 = (const float*)d_in[8];
  const float* l2r_w2 = (const float*)d_in[9];
  const float* l2r_b2 = (const float*)d_in[10];
  const float* l1r_w1 = (const float*)d_in[11];
  const float* l1r_b1 = (const float*)d_in[12];
  const float* l1r_w2 = (const float*)d_in[13];
  const float* l1r_b2 = (const float*)d_in[14];
  const float* emb    = (const float*)d_in[15];   // [3,64]
  const float* pos    = (const float*)d_in[16];   // [25,64]
  const AttnP pS8 = load_attn(d_in, 17);
  const AttnP pL2 = load_attn(d_in, 29);
  const AttnP pL1 = load_attn(d_in, 41);
  const float* out_w1 = (const float*)d_in[53];   // [128,64]
  const float* out_b1 = (const float*)d_in[54];
  const float* out_w2 = (const float*)d_in[55];   // [1,128]
  const float* out_b2 = (const float*)d_in[56];

  // -------- workspace layout (fp32) ----------------------------------------
  float* wsf = (float*)d_ws;
  size_t off = 0;
  auto alloc = [&](size_t n) { float* p = wsf + off; off += n; return p; };
  float* featS8 = alloc((size_t)N8 * 64);
  float* featL2 = alloc((size_t)N8 * 64);
  float* featL1 = alloc((size_t)N4 * 64);
  float* X8     = alloc((size_t)N8 * 64);
  float* X4     = alloc((size_t)N4 * 64);
  Scratch ws;
  ws.S1   = alloc((size_t)N4 * 64);
  ws.S2   = alloc((size_t)N4 * 64);
  ws.Km   = alloc((size_t)N4 * 64);
  ws.Vm   = alloc((size_t)N4 * 64);
  ws.MID  = alloc((size_t)N4 * 256);
  ws.kpos = alloc(25 * 64);
  ws.vpos = alloc(25 * 64);
  float* Y1 = alloc((size_t)N4);

  // -------- RCUs (NCHW in -> [pix,64] rows out) ----------------------------
  rcu(s8, s8r_w1, s8r_b1, s8r_w2, s8r_b2, ws.MID, featS8, Bn, H8, W8, stream);
  rcu(L2, l2r_w1, l2r_b1, l2r_w2, l2r_b2, ws.MID, featL2, Bn, H8, W8, stream);
  rcu(L1, l1r_w1, l1r_b1, l1r_w2, l1r_b2, ws.MID, featL1, Bn, H4, W4, stream);

  // token state x starts as s8 features
  hipMemcpyAsync(X8, featS8, (size_t)N8 * 64 * sizeof(float),
                 hipMemcpyDeviceToDevice, stream);

  // -------- cross-attention blocks -----------------------------------------
  cross_block(featS8, Bn, H8, W8, X8, pS8, pos, emb + 0 * 64, ws, stream);
  cross_block(featL2, Bn, H8, W8, X8, pL2, pos, emb + 1 * 64, ws, stream);

  // upsample tokens 48x64 -> 96x128 (align_corners)
  {
    const long tot = (long)N4 * 64;
    resize_rows_kernel<<<(tot + 255) / 256, 256, 0, stream>>>(
        X8, X4, Bn, H8, W8, H4, W4, 64);
  }

  cross_block(featL1, Bn, H4, W4, X4, pL1, pos, emb + 2 * 64, ws, stream);

  // -------- output head -----------------------------------------------------
  gemm_wmma_kernel<<<dim3(N4 / 32, 2), 256, 0, stream>>>(
      X4, out_w1, out_b1, nullptr, ws.MID, N4, 128, 64, 1);
  head_final_kernel<<<N4 / 8, 256, 0, stream>>>(ws.MID, out_w2, out_b2, Y1, N4);

  // softplus map [B,1,96,128] -> bilinear [B,1,384,512]
  {
    const long tot = (long)Bn * 384 * 512;
    resize_rows_kernel<<<(tot + 255) / 256, 256, 0, stream>>>(
        Y1, (float*)d_out, Bn, H4, W4, 384, 512, 1);
  }
}
